// LSTMModel_65575560675577
// MI455X (gfx1250) — matrix operand order repaired and checked by reading, not compiled, not run on hardware
//
#include <hip/hip_runtime.h>
#include <hip/hip_bf16.h>

typedef __attribute__((ext_vector_type(16))) _Float16 v16h;
typedef __attribute__((ext_vector_type(8)))  _Float16 v8h;
typedef __attribute__((ext_vector_type(4)))  _Float16 v4h;
typedef __attribute__((ext_vector_type(8)))  float    v8f;

typedef const __attribute__((address_space(1))) _Float16* gchp;
typedef const __attribute__((address_space(1))) v8h*      gv8hp;

// Problem constants
#define Bsz 256
#define Tt  512
#define Dd  64
#define Hh  256
#define Gg  1024          // 4*H
#define K0  320           // D + H   (layer 0 concat K)
#define K1  512           // H + H   (layer 1 concat K)
#define MR  32            // batch rows per workgroup (2 WMMA M-tiles)

// ---- fast activations (overflow-safe, raw v_rcp_f32) ----
static __device__ __forceinline__ float sigm(float v) {
    return __builtin_amdgcn_rcpf(1.0f + __expf(-v));
}
static __device__ __forceinline__ float tanh_fast(float x) {
    float ax = fabsf(x);
    float e  = __expf(2.0f * ax);                       // inf for large ax -> r -> 1
    float r  = 1.0f - 2.0f * __builtin_amdgcn_rcpf(e + 1.0f);
    return copysignf(r, x);
}

// Launder a kernel-arg pointer so LICM can't hoist loads out of the t-loop,
// but keep it in the GLOBAL address space (global_load_*, LOADcnt only).
static __device__ __forceinline__ gchp launder_global(const _Float16* p) {
    uintptr_t u = (uintptr_t)p;
    asm volatile("" : "+s"(u));
    return (gchp)u;
}

// A fragment (16x32 f16, M=lane%16): elements 0..7 -> K+0..7, 8..15 -> K+16..23,
// lanes>=16 shifted by +8 in K (caller folds +8 into the row pointer).
static __device__ __forceinline__ v16h ldA(const _Float16* p) {
    v8h lo = *(const v8h*)(p);
    v8h hi = *(const v8h*)(p + 16);
    return __builtin_shufflevector(lo, hi, 0,1,2,3,4,5,6,7,8,9,10,11,12,13,14,15);
}

// B fragment from the fragment-major packed weights: 16 contiguous f16 per lane.
static __device__ __forceinline__ v16h ldB(gchp p) {
    v8h lo = *(gv8hp)(p);
    v8h hi = *(gv8hp)(p + 8);
    return __builtin_shufflevector(lo, hi, 0,1,2,3,4,5,6,7,8,9,10,11,12,13,14,15);
}

#define WMMA(A, B, C) __builtin_amdgcn_wmma_f32_16x16x32_f16(false, A, false, B, (short)0, C, false, false)

// Packed fragment address: [kstep][coltile(64)][lane(32)][16 f16]
static __device__ __forceinline__ gchp frag_ptr(gchp wtp, int ks, int ct, int lane) {
    return wtp + ((size_t)(ks * 64 + ct) << 9) + (lane << 4);
}

// Two M-tiles x four gate tiles, software-pipelined over K.
template<int KDIM>
static __device__ __forceinline__ void gemm8(const _Float16* __restrict__ Arow0,
                                             const _Float16* __restrict__ Arow1,
                                             gchp wtp, int wv, int lane, v8f acc[2][4]) {
    #pragma unroll
    for (int m = 0; m < 2; ++m)
        #pragma unroll
        for (int t = 0; t < 4; ++t) acc[m][t] = v8f{};

    v16h b0 = ldB(frag_ptr(wtp, 0,      wv, lane));
    v16h b1 = ldB(frag_ptr(wtp, 0, 16 + wv, lane));
    v16h b2 = ldB(frag_ptr(wtp, 0, 32 + wv, lane));
    v16h b3 = ldB(frag_ptr(wtp, 0, 48 + wv, lane));

    #pragma unroll
    for (int k = 0; k < KDIM; k += 32) {
        const int ks = k >> 5;
        v16h a0 = ldA(Arow0 + k);
        v16h a1 = ldA(Arow1 + k);
        v16h n0, n1, n2, n3;
        if (k + 32 < KDIM) {                      // prefetch next k-step's weights
            n0 = ldB(frag_ptr(wtp, ks + 1,      wv, lane));
            n1 = ldB(frag_ptr(wtp, ks + 1, 16 + wv, lane));
            n2 = ldB(frag_ptr(wtp, ks + 1, 32 + wv, lane));
            n3 = ldB(frag_ptr(wtp, ks + 1, 48 + wv, lane));
        }
        acc[0][0] = WMMA(a0, b0, acc[0][0]);  acc[1][0] = WMMA(a1, b0, acc[1][0]);
        acc[0][1] = WMMA(a0, b1, acc[0][1]);  acc[1][1] = WMMA(a1, b1, acc[1][1]);
        acc[0][2] = WMMA(a0, b2, acc[0][2]);  acc[1][2] = WMMA(a1, b2, acc[1][2]);
        acc[0][3] = WMMA(a0, b3, acc[0][3]);  acc[1][3] = WMMA(a1, b3, acc[1][3]);
        if (k + 32 < KDIM) { b0 = n0; b1 = n1; b2 = n2; b3 = n3; }
    }
}

// Prologue: repack weights into fragment-major f16 layout, fuse biases.
// Packed element j: e=j&15 (fragment element), lane=(j>>4)&31, ct=(j>>9)&63 (column tile),
// ks=j>>15 (K step). col=ct*16+(lane&15), k=ks*32+((e&8)<<1)+((lane>>4)<<3)+(e&7).
// Layer0 K concat: k<64 -> W_ih0[col][k], else W_hh0[col][k-64]     (K=320)
// Layer1 K concat: k<256 -> W_ih1[col][k], else W_hh1[col][k-256]   (K=512)
__global__ void lstm_prep_kernel(const float* __restrict__ Wih0, const float* __restrict__ Whh0,
                                 const float* __restrict__ bih0, const float* __restrict__ bhh0,
                                 const float* __restrict__ Wih1, const float* __restrict__ Whh1,
                                 const float* __restrict__ bih1, const float* __restrict__ bhh1,
                                 _Float16* __restrict__ wt0, _Float16* __restrict__ wt1,
                                 float* __restrict__ bias0, float* __restrict__ bias1) {
    int j = blockIdx.x * blockDim.x + threadIdx.x;
    int e = j & 15, lane = (j >> 4) & 31, ct = (j >> 9) & 63, ks = j >> 15;
    int col = ct * 16 + (lane & 15);
    int k   = ks * 32 + ((e & 8) << 1) + ((lane >> 4) << 3) + (e & 7);
    if (j < K1 * Gg) {
        float v = (k < Hh) ? Wih1[col * Hh + k] : Whh1[col * Hh + (k - Hh)];
        wt1[j] = (_Float16)v;
    }
    if (j < K0 * Gg) {
        float v = (k < Dd) ? Wih0[col * Dd + k] : Whh0[col * Hh + (k - Dd)];
        wt0[j] = (_Float16)v;
    }
    if (j < Gg) {
        bias0[j] = bih0[j] + bhh0[j];
        bias1[j] = bih1[j] + bhh1[j];
    }
}

// Persistent two-layer LSTM: one workgroup per 32-batch-row tile, full T recurrence.
__global__ __launch_bounds__(512) void
lstm_main_kernel(const float* __restrict__ x,
                 const _Float16* __restrict__ wt0_in, const _Float16* __restrict__ wt1_in,
                 const float* __restrict__ bias0, const float* __restrict__ bias1,
                 const float* __restrict__ fcW, const float* __restrict__ fcb,
                 float* __restrict__ out) {
    __shared__ __align__(16) _Float16 A0[MR * K0];   // [ x_t (K 0..63) | hA (K 64..319) ]
    __shared__ __align__(16) _Float16 A1[MR * K1];   // [ hA  (K 0..255) | hB (K 256..511) ]

    const int tid  = threadIdx.x;
    const int wv   = tid >> 5;          // wave 0..15 -> feature slice
    const int lane = tid & 31;
    const int lm   = lane & 15;
    const int lh   = lane >> 4;
    const int fb   = wv * 16;           // feature base of this wave
    const int g    = blockIdx.x;        // batch tile (32 rows)

    for (int i = tid; i < MR * K0; i += 512) A0[i] = (_Float16)0.0f;
    for (int i = tid; i < MR * K1; i += 512) A1[i] = (_Float16)0.0f;

    v8f cA[2] = {v8f{}, v8f{}};
    v8f cB[2] = {v8f{}, v8f{}};

    const float b0i = bias0[      fb + lm], b0f = bias0[256 + fb + lm];
    const float b0g = bias0[512 + fb + lm], b0o = bias0[768 + fb + lm];
    const float b1i = bias1[      fb + lm], b1f = bias1[256 + fb + lm];
    const float b1g = bias1[512 + fb + lm], b1o = bias1[768 + fb + lm];

    const _Float16* Ar0m0 = A0 + (     lm) * K0 + (lh << 3);
    const _Float16* Ar0m1 = A0 + (16 + lm) * K0 + (lh << 3);
    const _Float16* Ar1m0 = A1 + (     lm) * K1 + (lh << 3);
    const _Float16* Ar1m1 = A1 + (16 + lm) * K1 + (lh << 3);

    __syncthreads();

    for (int t = 0; t < Tt; ++t) {
        // Per-iteration laundering keeps the B loads inside the t-loop,
        // in the global address space.
        gchp wt0 = launder_global(wt0_in);
        gchp wt1 = launder_global(wt1_in);

        // ---- stage x_t tile (32 rows x 64 f32 -> f16 LDS), 4 elems/thread ----
        {
            int idx = tid * 4;
            int row = idx >> 6, d = idx & 63;
            const float4 p = *(const float4*)(x + (((size_t)(g * MR + row)) * Tt + t) * Dd + d);
            v4h hv; hv[0] = (_Float16)p.x; hv[1] = (_Float16)p.y;
                    hv[2] = (_Float16)p.z; hv[3] = (_Float16)p.w;
            *(v4h*)(&A0[row * K0 + d]) = hv;
        }
        __syncthreads();

        // ---- layer 0 gates: [x_t | hA] @ wt0 ----
        v8f acc[2][4];
        gemm8<K0>(Ar0m0, Ar0m1, wt0, wv, lane, acc);
        __syncthreads();                 // all reads of A0 done before h writes

        #pragma unroll
        for (int m = 0; m < 2; ++m) {
            #pragma unroll
            for (int r = 0; r < 8; ++r) {
                float iv = sigm(acc[m][0][r] + b0i);
                float fv = sigm(acc[m][1][r] + b0f);
                float gv = tanh_fast(acc[m][2][r] + b0g);
                float ov = sigm(acc[m][3][r] + b0o);
                float c  = fv * cA[m][r] + iv * gv;
                cA[m][r] = c;
                _Float16 hh = (_Float16)(ov * tanh_fast(c));
                int rowr = m * 16 + r + (lh << 3);
                A0[rowr * K0 + Dd + fb + lm] = hh;   // recurrent input, layer 0
                A1[rowr * K1 +      fb + lm] = hh;   // feed-forward input, layer 1
            }
        }
        __syncthreads();

        // ---- layer 1 gates: [hA_t | hB] @ wt1 ----
        gemm8<K1>(Ar1m0, Ar1m1, wt1, wv, lane, acc);
        __syncthreads();                 // all reads of A1 done before hB writes

        #pragma unroll
        for (int m = 0; m < 2; ++m) {
            #pragma unroll
            for (int r = 0; r < 8; ++r) {
                float iv = sigm(acc[m][0][r] + b1i);
                float fv = sigm(acc[m][1][r] + b1f);
                float gv = tanh_fast(acc[m][2][r] + b1g);
                float ov = sigm(acc[m][3][r] + b1o);
                float c  = fv * cB[m][r] + iv * gv;
                cB[m][r] = c;
                _Float16 hh = (_Float16)(ov * tanh_fast(c));
                int rowr = m * 16 + r + (lh << 3);
                A1[rowr * K1 + Hh + fb + lm] = hh;
            }
        }
        __syncthreads();
    }

    // ---- final FC: out[b,o] = hB[b,:] . fcW[o,:] + fcb[o] ----
    if (tid < MR * 4) {
        int bl = tid >> 2, o = tid & 3;
        float s = fcb[o];
        const _Float16* hrow = A1 + bl * K1 + Hh;
        for (int j = 0; j < Hh; ++j) s += (float)hrow[j] * fcW[o * Hh + j];
        out[(g * MR + bl) * 4 + o] = s;
    }
}

extern "C" void kernel_launch(void* const* d_in, const int* in_sizes, int n_in,
                              void* d_out, int out_size, void* d_ws, size_t ws_size,
                              hipStream_t stream) {
    const float* x    = (const float*)d_in[0];
    const float* Wih0 = (const float*)d_in[1];
    const float* Whh0 = (const float*)d_in[2];
    const float* bih0 = (const float*)d_in[3];
    const float* bhh0 = (const float*)d_in[4];
    const float* Wih1 = (const float*)d_in[5];
    const float* Whh1 = (const float*)d_in[6];
    const float* bih1 = (const float*)d_in[7];
    const float* bhh1 = (const float*)d_in[8];
    const float* fcW  = (const float*)d_in[9];
    const float* fcb  = (const float*)d_in[10];

    // workspace: wt0 (320*1024 f16, packed) | wt1 (512*1024 f16, packed) | bias0 | bias1
    _Float16* wt0 = (_Float16*)d_ws;
    _Float16* wt1 = wt0 + (size_t)K0 * Gg;
    float* bias0  = (float*)(wt1 + (size_t)K1 * Gg);
    float* bias1  = bias0 + Gg;

    lstm_prep_kernel<<<(K1 * Gg + 255) / 256, 256, 0, stream>>>(
        Wih0, Whh0, bih0, bhh0, Wih1, Whh1, bih1, bhh1, wt0, wt1, bias0, bias1);

    lstm_main_kernel<<<Bsz / MR, 512, 0, stream>>>(
        x, wt0, wt1, bias0, bias1, fcW, fcb, (float*)d_out);
}
